// ResidualSparseBlock2D_12790412608007
// MI455X (gfx1250) — compile-verified, matrix operand order checked by reading, not run on hardware
//
#include <hip/hip_runtime.h>
#include <stdint.h>

// ---------------------------------------------------------------------------
// ResidualSparseBlock2D for MI455X (gfx1250, wave32, WMMA)
//
// Pipeline (all on `stream`):
//   pack W1,W2 -> bf16 hi/lo in WMMA B-operand lane layout
//   conv1 (bf16x3 WMMA, fp32 accum) -> t1, per-block BN partials
//   reduce + finalize -> scale1/shift1
//   conv2 (applies BN1+ReLU on gathered rows) -> t2, partials
//   reduce + finalize -> scale2/shift2
//   fused BN2 + residual + ReLU -> out
// ---------------------------------------------------------------------------

typedef __attribute__((ext_vector_type(16))) __bf16 v16bf;
typedef __attribute__((ext_vector_type(8)))  float  v8f;
typedef __attribute__((ext_vector_type(4)))  float  f4;

#define KTAPS 9
#define CCH   64
#define ROWS_PER_BLOCK 128
#define THREADS 256
// packed-B elements per weight tensor: taps * kChunks * colTiles * lanes * 16
#define BPACK_ELEMS (KTAPS * 2 * 4 * 32 * 16)   // 36864

// ---------------------------------------------------------------------------
// Pack W[k][cin][cout] (fp32) into per-lane WMMA-B bf16 layout, hi/lo split.
// For 16-bit operands, lane L (col n = 16t + (L&15)) holds element e with
//   K(e) = 32c + (e>=8 ? 16 : 0) + ((L&16)?8:0) + (e&7)
// so the conv kernel loads B as one contiguous 32B v16bf per lane.
// ---------------------------------------------------------------------------
__global__ __launch_bounds__(256)
void pack_weights_kernel(const float* __restrict__ W,
                         __bf16* __restrict__ Bhi, __bf16* __restrict__ Blo) {
  int idx = blockIdx.x * blockDim.x + threadIdx.x;
  if (idx >= BPACK_ELEMS) return;
  int e    = idx & 15;
  int lane = (idx >> 4) & 31;
  int t    = (idx >> 9) & 3;
  int c    = (idx >> 11) & 1;
  int tap  = idx >> 12;
  int khalf = (lane & 16) ? 8 : 0;
  int kk = 32 * c + ((e >= 8) ? 16 : 0) + khalf + (e & 7);
  int n  = 16 * t + (lane & 15);
  float wv = W[tap * CCH * CCH + kk * CCH + n];
  __bf16 h = (__bf16)wv;
  Bhi[idx] = h;
  Blo[idx] = (__bf16)(wv - (float)h);
}

// ---------------------------------------------------------------------------
// Sparse conv: out[i] = sum_k gather(X, nbr[i,k]) @ W[k], bf16x3 WMMA.
// Each wave: 16 rows x 64 cols (4 tiles of 16x16), 9 taps * 2 K-chunks.
// Writes raw conv output T and deterministic per-block BN partial sums.
// ---------------------------------------------------------------------------
template <bool APPLY_BN>
__global__ __launch_bounds__(THREADS)
void sparse_conv_kernel(const float* __restrict__ X,
                        const int*   __restrict__ nbr,
                        const __bf16* __restrict__ Bhi,
                        const __bf16* __restrict__ Blo,
                        const float* __restrict__ scIn,
                        const float* __restrict__ shIn,
                        float* __restrict__ T,
                        float* __restrict__ partial,
                        int N) {
  __shared__ float redS[16 * 64];
  __shared__ float redQ[16 * 64];

  const int lane   = threadIdx.x & 31;
  const int wave   = threadIdx.x >> 5;
  const int laneHi = (lane >> 4) & 1;
  const int khalf  = laneHi ? 8 : 0;
  const int rowBase = blockIdx.x * ROWS_PER_BLOCK;
  const int myRow   = rowBase + wave * 16 + (lane & 15);
  const bool valid  = myRow < N;
  const int  rClamp = valid ? myRow : 0;

  // Per-lane BN1 affine constants (channel set per lane is tap-invariant).
  float SC[2][16], SH[2][16];
  if constexpr (APPLY_BN) {
#pragma unroll
    for (int c = 0; c < 2; ++c)
#pragma unroll
      for (int e = 0; e < 16; ++e) {
        int ch = 32 * c + ((e >= 8) ? 16 : 0) + khalf + (e & 7);
        SC[c][e] = scIn[ch];
        SH[c][e] = shIn[ch];
      }
  }

  v8f acc[4];
#pragma unroll
  for (int t = 0; t < 4; ++t)
#pragma unroll
    for (int v = 0; v < 8; ++v) acc[t][v] = 0.0f;

#pragma unroll 1
  for (int tap = 0; tap < KTAPS; ++tap) {
    int nb = nbr[rClamp * KTAPS + tap];            // lanes L, L+16 share a row
    const float* rowp = X + (size_t)nb * CCH;
#pragma unroll
    for (int c = 0; c < 2; ++c) {
      const int cb = 32 * c + khalf;
      // A-operand lane layout = two contiguous 8-float stretches per lane.
      f4 q0 = *(const f4*)(rowp + cb);
      f4 q1 = *(const f4*)(rowp + cb + 4);
      f4 q2 = *(const f4*)(rowp + cb + 16);
      f4 q3 = *(const f4*)(rowp + cb + 20);
      float xs[16];
#pragma unroll
      for (int j = 0; j < 4; ++j) {
        xs[j] = q0[j]; xs[4 + j] = q1[j]; xs[8 + j] = q2[j]; xs[12 + j] = q3[j];
      }
      v16bf ah, al;
#pragma unroll
      for (int e = 0; e < 16; ++e) {
        float x = xs[e];
        if constexpr (APPLY_BN) x = fmaxf(fmaf(x, SC[c][e], SH[c][e]), 0.0f);
        x = valid ? x : 0.0f;                      // value-mask, no branch:
        __bf16 h = (__bf16)x;                      // EXEC stays all-ones for WMMA
        ah[e] = h;
        al[e] = (__bf16)(x - (float)h);
      }
      const __bf16* bb = Bhi + (size_t)((tap * 2 + c) * 4) * 32 * 16;
      const __bf16* lb = Blo + (size_t)((tap * 2 + c) * 4) * 32 * 16;
#pragma unroll
      for (int t = 0; t < 4; ++t) {
        v16bf bh = *(const v16bf*)(bb + (t * 32 + lane) * 16);
        v16bf bl = *(const v16bf*)(lb + (t * 32 + lane) * 16);
        // bf16x3: hi*hi + hi*lo + lo*hi (fp32 accumulate) ~ fp32 accuracy
        acc[t] = __builtin_amdgcn_wmma_f32_16x16x32_bf16(
            false, ah, false, bh, (short)0, acc[t], false, false);
        acc[t] = __builtin_amdgcn_wmma_f32_16x16x32_bf16(
            false, ah, false, bl, (short)0, acc[t], false, false);
        acc[t] = __builtin_amdgcn_wmma_f32_16x16x32_bf16(
            false, al, false, bh, (short)0, acc[t], false, false);
      }
    }
  }

  // ---- deterministic BN partial sums (no atomics): each of the 1024
  // (wave, laneHi, channel) LDS slots is written exactly once.
#pragma unroll
  for (int t = 0; t < 4; ++t) {
    float s = 0.0f, q = 0.0f;
#pragma unroll
    for (int v = 0; v < 8; ++v) { float x = acc[t][v]; s += x; q += x * x; }
    int ch = 16 * t + (lane & 15);
    redS[(wave * 2 + laneHi) * 64 + ch] = s;
    redQ[(wave * 2 + laneHi) * 64 + ch] = q;
  }
  __syncthreads();
  if (threadIdx.x < 64) {
    float s = 0.0f, q = 0.0f;
#pragma unroll 1
    for (int i = 0; i < 16; ++i) {
      s += redS[i * 64 + threadIdx.x];
      q += redQ[i * 64 + threadIdx.x];
    }
    partial[(size_t)blockIdx.x * 128 + threadIdx.x]      = s;
    partial[(size_t)blockIdx.x * 128 + 64 + threadIdx.x] = q;
  }

  // ---- store raw conv output (C/D layout: VGPR v -> row M = v + 8*laneHi)
#pragma unroll
  for (int t = 0; t < 4; ++t) {
#pragma unroll
    for (int v = 0; v < 8; ++v) {
      int r = rowBase + wave * 16 + v + (laneHi ? 8 : 0);
      if (r < N) T[(size_t)r * CCH + 16 * t + (lane & 15)] = acc[t][v];
    }
  }
}

// ---------------------------------------------------------------------------
// Deterministic two-stage reduction of per-block partials -> scale/shift.
// ---------------------------------------------------------------------------
__global__ __launch_bounds__(128)
void reduce_stage1(const float* __restrict__ partial, float* __restrict__ p2,
                   int nBlocks) {
  int j = threadIdx.x;          // 0..63 sums, 64..127 sumsqs (same stride)
  int b = blockIdx.x;           // 64 blocks
  int per = (nBlocks + 63) >> 6;
  int lo = b * per;
  int hi = lo + per; if (hi > nBlocks) hi = nBlocks;
  float s = 0.0f;
  for (int blk = lo; blk < hi; ++blk) s += partial[(size_t)blk * 128 + j];
  p2[b * 128 + j] = s;
}

__global__ __launch_bounds__(64)
void bn_finalize(const float* __restrict__ p2,
                 const float* __restrict__ gamma, const float* __restrict__ beta,
                 float* __restrict__ scale, float* __restrict__ shift,
                 float invN) {
  int ch = threadIdx.x;
  float s = 0.0f, q = 0.0f;
  for (int b = 0; b < 64; ++b) {
    s += p2[b * 128 + ch];
    q += p2[b * 128 + 64 + ch];
  }
  float mean = s * invN;
  float var  = q * invN - mean * mean;
  float rs   = rsqrtf(var + 1e-5f);
  float sc   = gamma[ch] * rs;
  scale[ch] = sc;
  shift[ch] = beta[ch] - mean * sc;
}

// ---------------------------------------------------------------------------
// out = relu(bn2(t2) + feats), float4-vectorized (pure bandwidth).
// ---------------------------------------------------------------------------
__global__ __launch_bounds__(256)
void fuse_out_kernel(const float* __restrict__ t2, const float* __restrict__ feats,
                     const float* __restrict__ scale, const float* __restrict__ shift,
                     float* __restrict__ out, long long nVec) {
  long long i = (long long)blockIdx.x * blockDim.x + threadIdx.x;
  if (i >= nVec) return;
  int part = (int)(i & 15);                 // 64 ch = 16 float4 groups/row
  f4 sc = *(const f4*)(scale + part * 4);
  f4 sh = *(const f4*)(shift + part * 4);
  f4 a = ((const f4*)t2)[i];
  f4 r = ((const f4*)feats)[i];
  f4 o;
#pragma unroll
  for (int j = 0; j < 4; ++j)
    o[j] = fmaxf(fmaf(a[j], sc[j], sh[j]) + r[j], 0.0f);
  ((f4*)out)[i] = o;
}

// ---------------------------------------------------------------------------
extern "C" void kernel_launch(void* const* d_in, const int* in_sizes, int n_in,
                              void* d_out, int out_size, void* d_ws, size_t ws_size,
                              hipStream_t stream) {
  const float* feats = (const float*)d_in[0];
  const float* W1    = (const float*)d_in[1];
  const float* g1    = (const float*)d_in[2];
  const float* b1    = (const float*)d_in[3];
  const float* W2    = (const float*)d_in[4];
  const float* g2    = (const float*)d_in[5];
  const float* b2    = (const float*)d_in[6];
  const int*   nbr1  = (const int*)d_in[7];
  const int*   nbr2  = (const int*)d_in[8];
  float* out = (float*)d_out;

  const int N = in_sizes[0] / CCH;
  const int nBlocks = (N + ROWS_PER_BLOCK - 1) / ROWS_PER_BLOCK;

  // Workspace carve-out (256B aligned): t1, t2, packed weights, partials.
  char* p = (char*)d_ws;
  auto take = [&](size_t bytes) -> char* {
    char* r = p;
    p += (bytes + 255) & ~(size_t)255;
    return r;
  };
  float*  t1      = (float*)take((size_t)N * CCH * sizeof(float));
  float*  t2      = (float*)take((size_t)N * CCH * sizeof(float));
  __bf16* W1hi    = (__bf16*)take((size_t)BPACK_ELEMS * 2);
  __bf16* W1lo    = (__bf16*)take((size_t)BPACK_ELEMS * 2);
  __bf16* W2hi    = (__bf16*)take((size_t)BPACK_ELEMS * 2);
  __bf16* W2lo    = (__bf16*)take((size_t)BPACK_ELEMS * 2);
  float*  partial = (float*)take((size_t)nBlocks * 128 * sizeof(float));
  float*  p2      = (float*)take((size_t)64 * 128 * sizeof(float));
  float*  sc1     = (float*)take(64 * sizeof(float));
  float*  sh1     = (float*)take(64 * sizeof(float));
  float*  sc2     = (float*)take(64 * sizeof(float));
  float*  sh2     = (float*)take(64 * sizeof(float));

  const int packGrid = (BPACK_ELEMS + 255) / 256;
  pack_weights_kernel<<<packGrid, 256, 0, stream>>>(W1, W1hi, W1lo);
  pack_weights_kernel<<<packGrid, 256, 0, stream>>>(W2, W2hi, W2lo);

  // conv1 -> t1 (+ BN1 partials)
  sparse_conv_kernel<false><<<nBlocks, THREADS, 0, stream>>>(
      feats, nbr1, W1hi, W1lo, nullptr, nullptr, t1, partial, N);
  reduce_stage1<<<64, 128, 0, stream>>>(partial, p2, nBlocks);
  bn_finalize<<<1, 64, 0, stream>>>(p2, g1, b1, sc1, sh1, 1.0f / (float)N);

  // conv2 with lazy BN1+ReLU on gathered rows -> t2 (+ BN2 partials)
  sparse_conv_kernel<true><<<nBlocks, THREADS, 0, stream>>>(
      t1, nbr2, W2hi, W2lo, sc1, sh1, t2, partial, N);
  reduce_stage1<<<64, 128, 0, stream>>>(partial, p2, nBlocks);
  bn_finalize<<<1, 64, 0, stream>>>(p2, g2, b2, sc2, sh2, 1.0f / (float)N);

  // fused BN2 + residual + ReLU
  long long nVec = (long long)N * (CCH / 4);
  long long fb = (nVec + 255) / 256;
  fuse_out_kernel<<<(unsigned)fb, 256, 0, stream>>>(t2, feats, sc2, sh2, out, nVec);
}
